// RNNModel_4269197492318
// MI455X (gfx1250) — matrix operand order repaired and checked by reading, not compiled, not run on hardware
//
#include <hip/hip_runtime.h>
#include <math.h>

// ---------------------------------------------------------------- types ----
typedef __attribute__((ext_vector_type(16))) __bf16        v16bf;
typedef __attribute__((ext_vector_type(8)))  float         v8f;
typedef __attribute__((ext_vector_type(8)))  unsigned int  u32x8;

struct U128x2 { uint4 lo, hi; };

__device__ __forceinline__ unsigned short f2bf(float f) {
  unsigned int u = __builtin_bit_cast(unsigned int, f);
  u = (u + 0x7FFFu + ((u >> 16) & 1u)) >> 16;   // round-to-nearest-even
  return (unsigned short)u;
}
__device__ __forceinline__ float bf2f(unsigned short h) {
  unsigned int u = ((unsigned int)h) << 16;
  return __builtin_bit_cast(float, u);
}

// ------------------------------------------------------------- constants ---
#define N_MELS 128
#define HID    512
#define TWO_H  1024
#define G3     1536
#define BATCH  4
#define FR     8
#define NF     1024          // TLEN / FR
#define SEQ    4096          // BATCH * NF   (rnn1 batch)
#define ROWS1  32768         // SEQ * FR
#define OUTN   61

// GEMM tiling
#define BM 128
#define BN 128
#define BK 32
#define LDT 40               // padded ushort stride for LDS tiles
#define HPAD 520             // padded ushort stride for GRU h in LDS

// -------------------------------------------------- WMMA fragment loaders --
// A fragment: 16(M) x 32(K) bf16. lane = M (mod 16); lane-half selects K
// contiguous (matches B fragments): half 0 -> K 0..15, half 1 -> K 16..31.
__device__ __forceinline__ v16bf lds_a_frag(const unsigned short* tile, int ldt) {
  int lane = threadIdx.x & 31;
  int m = lane & 15, half = lane >> 4;
  const unsigned short* rp = tile + m * ldt;
  u32x8 u;
#pragma unroll
  for (int p = 0; p < 8; ++p) {
    int k0 = half * 16 + (p << 1); // half*16+p*2
    u[p] = *(const unsigned int*)(rp + k0);              // packs K=k0,k0+1
  }
  return __builtin_bit_cast(v16bf, u);
}

// B fragment: 32(K) x 16(N) bf16 taken from weight tile stored as N x K rows.
// lane = N (mod 16); lane-half selects K 0..15 / 16..31 (contiguous in W row).
__device__ __forceinline__ v16bf lds_b_frag(const unsigned short* tile, int n0, int ldt) {
  int lane = threadIdx.x & 31;
  int n = lane & 15, half = lane >> 4;
  const unsigned short* rp = tile + (n0 + n) * ldt + half * 16;
  U128x2 u;
  u.lo = *(const uint4*)(rp);
  u.hi = *(const uint4*)(rp + 8);
  return __builtin_bit_cast(v16bf, u);
}

// B fragment directly from global weight matrix W (N x K row-major, bf16).
__device__ __forceinline__ v16bf glb_b_frag(const unsigned short* __restrict__ W,
                                            int n0, int ldb, int k0) {
  int lane = threadIdx.x & 31;
  int n = lane & 15, half = lane >> 4;
  const unsigned short* rp = W + (long)(n0 + n) * ldb + k0 + half * 16;
  U128x2 u;
  u.lo = *(const uint4*)(rp);
  u.hi = *(const uint4*)(rp + 8);
  return __builtin_bit_cast(v16bf, u);
}

// --------------------------------------------- async tile staging (CDNA5) --
// Copies one BMxBK (or BNxBK) bf16 tile into LDS with
// GLOBAL_LOAD_ASYNC_TO_LDS_B128 (ASYNCcnt-tracked, no VGPR staging).
// Out-of-range rows are zero-filled with a plain ds_store (disjoint rows, so
// no overlap with the async writes).
__device__ __forceinline__ void stage_tile_async(const unsigned short* __restrict__ gbase,
                                                 long ld, long row0, long rowMax,
                                                 int k0, unsigned short* lbuf) {
#pragma unroll
  for (int it = 0; it < 2; ++it) {
    int c = threadIdx.x + it * 256;
    int row = c >> 2, col = (c & 3) * 8;
    unsigned short* lp = &lbuf[row * LDT + col];
    long gr = row0 + row;
    if (gr < rowMax) {
      unsigned int laddr = (unsigned int)(unsigned long long)lp;
      const unsigned short* gp = gbase + gr * ld + k0 + col;
      asm volatile("global_load_async_to_lds_b128 %0, %1, off"
                   :: "v"(laddr), "v"(gp) : "memory");
    } else {
      *(uint4*)lp = make_uint4(0u, 0u, 0u, 0u);
    }
  }
}

__device__ __forceinline__ void wait_async_zero() {
  asm volatile("s_wait_asynccnt 0x0" ::: "memory");
}

// ------------------------------------------------------- utility kernels ---
__global__ void cvt_f32_bf16_kernel(const float* __restrict__ s,
                                    unsigned short* __restrict__ d, long n) {
  long i = (long)blockIdx.x * blockDim.x + threadIdx.x;
  long stride = (long)gridDim.x * blockDim.x;
  for (; i < n; i += stride) d[i] = f2bf(s[i]);
}

__global__ void zero_bf16_kernel(unsigned short* __restrict__ d, long n) {
  long i = (long)blockIdx.x * blockDim.x + threadIdx.x;
  long stride = (long)gridDim.x * blockDim.x;
  for (; i < n; i += stride) d[i] = 0;
}

// ------------------------------------------------------------ GEMM kernel --
// C = act( A(bf16, MxK) @ W^T(bf16, NxK) + bias + extra )
// act: 0 = identity, 1 = tanh. Any of Cf / Cb1 / Cb2 may be null.
// Double-buffered LDS pipeline fed by async-to-LDS copies:
//   iter k: s_wait_asynccnt 0 ; barrier ; issue tile k+1 ; wmma on tile k
__global__ __launch_bounds__(256)
void wmma_gemm_kernel(const unsigned short* __restrict__ A, long lda,
                      const unsigned short* __restrict__ W,   // N x K
                      const float* __restrict__ bias,
                      const float* __restrict__ extra, long eld,
                      float* __restrict__ Cf, long ldcf,
                      unsigned short* __restrict__ Cb1, long ldc1,
                      unsigned short* __restrict__ Cb2, long ldc2,
                      int M, int N, int K, int act) {
  __shared__ unsigned short As[2][BM * LDT];
  __shared__ unsigned short Bs[2][BN * LDT];

  const int tid = threadIdx.x;
  const int wid = tid >> 5;
  const int wm = wid >> 1;          // 0..3  -> 32-row slice
  const int wn = wid & 1;           // 0..1  -> 64-col slice
  const long blockM = (long)blockIdx.y * BM;
  const long blockN = (long)blockIdx.x * BN;

  v8f acc[2][4] = {};

  const int nk = K / BK;            // K is a multiple of 32 in all our calls
  // prologue: stage tile 0 into buffer 0
  stage_tile_async(A, lda, blockM, M, 0, As[0]);
  stage_tile_async(W, (long)K, blockN, N, 0, Bs[0]);

  for (int kt = 0; kt < nk; ++kt) {
    wait_async_zero();              // tile kt landed (this wave's copies)
    __syncthreads();                // whole tile visible to all waves
    if (kt + 1 < nk) {              // overlap: fetch next tile into alt buffer
      stage_tile_async(A, lda, blockM, M, (kt + 1) * BK, As[(kt + 1) & 1]);
      stage_tile_async(W, (long)K, blockN, N, (kt + 1) * BK, Bs[(kt + 1) & 1]);
    }

    const unsigned short* at = As[kt & 1];
    const unsigned short* bt = Bs[kt & 1];
    v16bf af[2], bf[4];
#pragma unroll
    for (int i = 0; i < 2; ++i)
      af[i] = lds_a_frag(&at[(wm * 32 + i * 16) * LDT], LDT);
#pragma unroll
    for (int j = 0; j < 4; ++j)
      bf[j] = lds_b_frag(bt, wn * 64 + j * 16, LDT);
#pragma unroll
    for (int i = 0; i < 2; ++i)
#pragma unroll
      for (int j = 0; j < 4; ++j)
        acc[i][j] = __builtin_amdgcn_wmma_f32_16x16x32_bf16(
            false, af[i], false, bf[j], (short)0, acc[i][j], false, false);
  }

  // epilogue: C layout -> element v of lane = (row v + 8*(lane>=16), col lane&15)
  const int lane = tid & 31;
  const int half = lane >> 4, n16 = lane & 15;
#pragma unroll
  for (int i = 0; i < 2; ++i) {
#pragma unroll
    for (int j = 0; j < 4; ++j) {
#pragma unroll
      for (int v = 0; v < 8; ++v) {
        long gm = blockM + wm * 32 + i * 16 + v + 8 * half;
        long gn = blockN + wn * 64 + j * 16 + n16;
        if (gm < M && gn < N) {
          float val = acc[i][j][v];
          if (bias)  val += bias[gn];
          if (extra) val += extra[gm * eld + gn];
          if (act == 1) val = tanhf(val);
          if (Cf)  Cf[gm * ldcf + gn] = val;
          if (Cb1) Cb1[gm * ldc1 + gn] = f2bf(val);
          if (Cb2) Cb2[gm * ldc2 + gn] = f2bf(val);
        }
      }
    }
  }
}

// --------------------------------------------------- persistent GRU scan ---
// grid = 2 (block 0 = forward, block 1 = backward), 512 threads = 16 waves.
// h (16 x 512, rows 4..15 = zero padding) lives in LDS; per step each wave
// computes gate columns [32w, 32w+32) for r, z, n via WMMA against Whh
// streamed from global (L2-resident), then does the gating in registers.
__global__ __launch_bounds__(512)
void gru_scan_kernel(const float* __restrict__ xw_f, const float* __restrict__ xw_b,
                     const unsigned short* __restrict__ Whh_f,
                     const unsigned short* __restrict__ Whh_b,
                     const float* __restrict__ bhh_f, const float* __restrict__ bhh_b,
                     unsigned short* __restrict__ yout, int nf) {
  __shared__ unsigned short hsm[16 * HPAD];

  const int dir = blockIdx.x;
  const float* __restrict__ xw = dir ? xw_b : xw_f;
  const unsigned short* __restrict__ W = dir ? Whh_b : Whh_f;
  const float* __restrict__ bh = dir ? bhh_b : bhh_f;
  const int coloff = dir * HID;

  const int tid = threadIdx.x;
  for (int i = tid; i < 16 * HPAD; i += blockDim.x) hsm[i] = 0;
  __syncthreads();

  const int wave = tid >> 5;
  const int col0 = wave * 32;            // hidden slice owned by this wave
  const int lane = tid & 31;
  const int half = lane >> 4, n16 = lane & 15;

  for (int s = 0; s < nf; ++s) {
    const int t = dir ? (nf - 1 - s) : s;

    v8f acc[3][2] = {};
#pragma unroll 1
    for (int kt = 0; kt < HID / BK; ++kt) {
      v16bf a = lds_a_frag(&hsm[kt * BK], HPAD);
#pragma unroll
      for (int g = 0; g < 3; ++g)
#pragma unroll
        for (int j = 0; j < 2; ++j) {
          v16bf b = glb_b_frag(W, g * HID + col0 + j * 16, HID, kt * BK);
          acc[g][j] = __builtin_amdgcn_wmma_f32_16x16x32_bf16(
              false, a, false, b, (short)0, acc[g][j], false, false);
        }
    }
    __syncthreads();   // all h reads done before h update

#pragma unroll
    for (int v = 0; v < 8; ++v) {
      int m = v + 8 * half;              // padded batch row
      if (m < BATCH) {
        long xrow = ((long)m * nf + t) * (long)G3;
#pragma unroll
        for (int j = 0; j < 2; ++j) {
          int col = col0 + j * 16 + n16;
          float ghr = acc[0][j][v] + bh[col];
          float ghz = acc[1][j][v] + bh[HID + col];
          float ghn = acc[2][j][v] + bh[2 * HID + col];
          float r = 1.0f / (1.0f + expf(-(xw[xrow + col] + ghr)));
          float z = 1.0f / (1.0f + expf(-(xw[xrow + HID + col] + ghz)));
          float n = tanhf(xw[xrow + 2 * HID + col] + r * ghn);
          float hold = bf2f(hsm[m * HPAD + col]);
          float hnew = (1.0f - z) * n + z * hold;
          unsigned short hb = f2bf(hnew);
          hsm[m * HPAD + col] = hb;      // exclusive columns: no cross-wave race
          yout[((long)m * nf + t) * (long)TWO_H + coloff + col] = hb;
        }
      }
    }
    __syncthreads();   // h update visible before next step's reads
  }
}

// ---------------------------------------------------------------- launch ---
extern "C" void kernel_launch(void* const* d_in, const int* in_sizes, int n_in,
                              void* d_out, int out_size, void* d_ws, size_t ws_size,
                              hipStream_t stream) {
  (void)in_sizes; (void)n_in; (void)out_size; (void)ws_size;

  const float* batch = (const float*)d_in[0];
  const float* fc1_w = (const float*)d_in[2];
  const float* fc1_b = (const float*)d_in[3];
  // rnn1: idx 4 + (l*2+d)*4 -> Wih, Whh, bih, bhh
  const float* fc2_w = (const float*)d_in[28];
  const float* fc2_b = (const float*)d_in[29];
  // rnn2: idx 30 + (l*2+d)*4
  const float* fc3_w = (const float*)d_in[54];
  const float* fc3_b = (const float*)d_in[55];
  float* out = (float*)d_out;

  // bump allocator over d_ws
  char* wsbase = (char*)d_ws;
  size_t off = 0;
  auto alloc = [&](size_t bytes) -> void* {
    off = (off + 255) & ~(size_t)255;
    void* p = wsbase + off;
    off += bytes;
    return p;
  };

  auto cvt = [&](const float* s, unsigned short* d, long n) {
    int blocks = (int)((n + 255) / 256);
    if (blocks > 8192) blocks = 8192;
    cvt_f32_bf16_kernel<<<blocks, 256, 0, stream>>>(s, d, n);
  };
  auto zero = [&](unsigned short* d, long n) {
    int blocks = (int)((n + 255) / 256);
    if (blocks > 8192) blocks = 8192;
    zero_bf16_kernel<<<blocks, 256, 0, stream>>>(d, n);
  };
  auto gemm = [&](const unsigned short* A, long lda, const unsigned short* W,
                  const float* bias, const float* extra, long eld,
                  float* Cf, long ldcf, unsigned short* C1, long ldc1,
                  unsigned short* C2, long ldc2, int M, int N, int K, int act) {
    dim3 g((N + BN - 1) / BN, (M + BM - 1) / BM);
    wmma_gemm_kernel<<<g, 256, 0, stream>>>(A, lda, W, bias, extra, eld,
                                            Cf, ldcf, C1, ldc1, C2, ldc2,
                                            M, N, K, act);
  };

  // ---- workspace ----
  unsigned short* batch_bf = (unsigned short*)alloc((size_t)ROWS1 * N_MELS * 2);
  unsigned short* wfc1 = (unsigned short*)alloc((size_t)N_MELS * N_MELS * 2);
  unsigned short* wih1[6]; unsigned short* whh1[6];
  for (int l = 0; l < 3; ++l) {
    int ind = l ? TWO_H : N_MELS;
    for (int d = 0; d < 2; ++d) {
      wih1[l * 2 + d] = (unsigned short*)alloc((size_t)HID * ind * 2);
      whh1[l * 2 + d] = (unsigned short*)alloc((size_t)HID * HID * 2);
    }
  }
  unsigned short* wfc2 = (unsigned short*)alloc((size_t)TWO_H * TWO_H * 2);
  unsigned short* wih2[6]; unsigned short* whh2[6];
  for (int i = 0; i < 6; ++i) {
    wih2[i] = (unsigned short*)alloc((size_t)G3 * TWO_H * 2);
    whh2[i] = (unsigned short*)alloc((size_t)G3 * HID * 2);
  }
  unsigned short* wfc3 = (unsigned short*)alloc((size_t)OUTN * TWO_H * 2);

  unsigned short* x0 = (unsigned short*)alloc((size_t)ROWS1 * N_MELS * 2);
  unsigned short* xa = (unsigned short*)alloc((size_t)ROWS1 * TWO_H * 2);
  unsigned short* xb = (unsigned short*)alloc((size_t)ROWS1 * TWO_H * 2);
  float* xw1[2];
  xw1[0] = (float*)alloc((size_t)ROWS1 * HID * 4);
  xw1[1] = (float*)alloc((size_t)ROWS1 * HID * 4);
  unsigned short* hbuf[2];
  hbuf[0] = (unsigned short*)alloc((size_t)SEQ * HID * 2);
  hbuf[1] = (unsigned short*)alloc((size_t)SEQ * HID * 2);
  unsigned short* y0 = (unsigned short*)alloc((size_t)SEQ * TWO_H * 2);
  float* xw2[2];
  xw2[0] = (float*)alloc((size_t)SEQ * G3 * 4);
  xw2[1] = (float*)alloc((size_t)SEQ * G3 * 4);
  unsigned short* ya = (unsigned short*)alloc((size_t)SEQ * TWO_H * 2);
  unsigned short* yb = (unsigned short*)alloc((size_t)SEQ * TWO_H * 2);

  // ---- convert inputs / weights to bf16 ----
  cvt(batch, batch_bf, (long)ROWS1 * N_MELS);
  cvt(fc1_w, wfc1, (long)N_MELS * N_MELS);
  for (int l = 0; l < 3; ++l) {
    int ind = l ? TWO_H : N_MELS;
    for (int d = 0; d < 2; ++d) {
      int base = 4 + (l * 2 + d) * 4;
      cvt((const float*)d_in[base + 0], wih1[l * 2 + d], (long)HID * ind);
      cvt((const float*)d_in[base + 1], whh1[l * 2 + d], (long)HID * HID);
    }
  }
  cvt(fc2_w, wfc2, (long)TWO_H * TWO_H);
  for (int i = 0; i < 6; ++i) {
    int base = 30 + i * 4;
    cvt((const float*)d_in[base + 0], wih2[i], (long)G3 * TWO_H);
    cvt((const float*)d_in[base + 1], whh2[i], (long)G3 * HID);
  }
  cvt(fc3_w, wfc3, (long)OUTN * TWO_H);

  // ---- fc1: x0 = batch @ fc1_w^T + b ----
  gemm(batch_bf, N_MELS, wfc1, fc1_b, nullptr, 0,
       nullptr, 0, x0, N_MELS, nullptr, 0, ROWS1, N_MELS, N_MELS, 0);

  // ---- rnn1: 3 bidirectional tanh layers ----
  unsigned short* in1 = x0;
  for (int l = 0; l < 3; ++l) {
    int ind = l ? TWO_H : N_MELS;
    unsigned short* out1 = (l & 1) ? xb : xa;   // l0->xa, l1->xb, l2->xa
    if (l == 2) out1 = xa;
    for (int d = 0; d < 2; ++d) {
      int base = 4 + (l * 2 + d) * 4;
      const float* bih = (const float*)d_in[base + 2];
      const float* bhh = (const float*)d_in[base + 3];
      // xw = X @ Wih^T + bih  (f32)
      gemm(in1, ind, wih1[l * 2 + d], bih, nullptr, 0,
           xw1[d], HID, nullptr, 0, nullptr, 0, ROWS1, HID, ind, 0);
      zero(hbuf[d], (long)SEQ * HID);
      for (int s = 0; s < FR; ++s) {
        int t = d ? (FR - 1 - s) : s;
        // h = tanh(h @ Whh^T + bhh + xw_t); write h (in place) and layer out
        gemm(hbuf[d], HID, whh1[l * 2 + d], bhh,
             xw1[d] + (long)t * HID, (long)FR * HID,
             nullptr, 0,
             hbuf[d], HID,
             out1 + (long)t * TWO_H + (long)d * HID, (long)FR * TWO_H,
             SEQ, HID, HID, 1);
      }
    }
    in1 = out1;
  }

  // ---- fc2 on last time-step (t = FR-1) of rnn1 output ----
  gemm(in1 + (long)(FR - 1) * TWO_H, (long)FR * TWO_H, wfc2, fc2_b, nullptr, 0,
       nullptr, 0, y0, TWO_H, nullptr, 0, SEQ, TWO_H, TWO_H, 0);

  // ---- rnn2: 3 bidirectional GRU layers ----
  unsigned short* in2 = y0;
  for (int l = 0; l < 3; ++l) {
    unsigned short* out2 = (l & 1) ? yb : ya;   // l0->ya, l1->yb, l2->ya
    if (l == 2) out2 = ya;
    for (int d = 0; d < 2; ++d) {
      int base = 30 + (l * 2 + d) * 4;
      const float* bih = (const float*)d_in[base + 2];
      gemm(in2, TWO_H, wih2[l * 2 + d], bih, nullptr, 0,
           xw2[d], G3, nullptr, 0, nullptr, 0, SEQ, G3, TWO_H, 0);
    }
    const float* bhh_f = (const float*)d_in[30 + (l * 2 + 0) * 4 + 3];
    const float* bhh_b = (const float*)d_in[30 + (l * 2 + 1) * 4 + 3];
    gru_scan_kernel<<<dim3(2), dim3(512), 0, stream>>>(
        xw2[0], xw2[1], whh2[l * 2 + 0], whh2[l * 2 + 1],
        bhh_f, bhh_b, out2, NF);
    in2 = out2;
  }

  // ---- fc3 -> d_out (f32) ----
  gemm(in2, TWO_H, wfc3, fc3_b, nullptr, 0,
       out, OUTN, nullptr, 0, nullptr, 0, SEQ, OUTN, TWO_H, 0);
}
